// Cross_SwinTransformerBlock_44822278701532
// MI455X (gfx1250) — matrix operand order repaired and checked by reading, not compiled, not run on hardware
//
#include <hip/hip_runtime.h>
#include <hip/hip_bf16.h>
#include <math.h>

typedef __attribute__((ext_vector_type(16))) _Float16 v16h;
typedef __attribute__((ext_vector_type(8)))  _Float16 v8h;
typedef __attribute__((ext_vector_type(8)))  float    v8f;
typedef __attribute__((ext_vector_type(4)))  unsigned int v4u;
typedef __attribute__((ext_vector_type(8)))  int      v8i;
typedef __attribute__((ext_vector_type(4)))  int      v4i;

#if defined(__has_builtin)
#if __has_builtin(__builtin_amdgcn_tensor_load_to_lds)
#define HAVE_TDM 1
#endif
#endif

#define WMMA_F32_F16(a,b,c) \
  __builtin_amdgcn_wmma_f32_16x16x32_f16(false,(a),false,(b),(short)0,(c),false,false)

static __device__ __forceinline__ int laneId(){ return threadIdx.x & 31; }

// A fragment: 16 rows x 32 K, row-major source, row stride ld (elements).
// ISA 7.12.2 16-bit A 16x32: lanes 0-15 hold K {0..7,16..23}; lanes 16-31 hold K {8..15,24..31}.
static __device__ __forceinline__ v16h load_frag_a(const _Float16* p, int ld){
  int l = laneId();
  int m  = l & 15;
  int kb = (l >> 4) << 3;           // 0 or 8
  v8h a0 = *reinterpret_cast<const v8h*>(p + m*ld + kb);
  v8h a1 = *reinterpret_cast<const v8h*>(p + m*ld + kb + 16);
  v16h r;
#pragma unroll
  for (int t=0;t<8;++t){ r[t]=a0[t]; r[8+t]=a1[t]; }
  return r;
}

// B fragment: 32 K x 16 N, sourced column-major via BT (N rows x K), row stride ld.
// Lane n<16 holds K {0..7,16..23}; lane>=16 holds K {8..15,24..31} (matches A layout).
static __device__ __forceinline__ v16h load_frag_b(const _Float16* pT, int ld){
  int l = laneId();
  int n  = l & 15;
  int kb = (l >> 4) << 3;           // 0 or 8
  v8h b0 = *reinterpret_cast<const v8h*>(pT + n*ld + kb);
  v8h b1 = *reinterpret_cast<const v8h*>(pT + n*ld + kb + 16);
  v16h r;
#pragma unroll
  for (int t=0;t<8;++t){ r[t]=b0[t]; r[8+t]=b1[t]; }
  return r;
}

// per-lane async global->LDS copy of one 16B chunk (ASYNCcnt path)
static __device__ __forceinline__ void async_copy_b128(unsigned int lds_off, const void* gaddr){
  unsigned long long ga = (unsigned long long)(uintptr_t)gaddr;
  asm volatile("global_load_async_to_lds_b128 %0, %1, off"
               :: "v"(lds_off), "v"(ga) : "memory");
}
static __device__ __forceinline__ void async_wait0(){
  asm volatile("s_wait_asynccnt 0x0" ::: "memory");
}

// ---- prep: fp32 weight (K x N row-major) -> f16 transposed (N x K) ----
__global__ void transpose_f16_kernel(const float* __restrict__ src, _Float16* __restrict__ dst,
                                     int K, int Nn){
  int idx = blockIdx.x * 256 + threadIdx.x;
  if (idx >= K*Nn) return;
  int n = idx / K;
  int k = idx - n*K;
  dst[idx] = (_Float16)src[(size_t)k*Nn + n];
}

// ---- LayerNorm (+optional fused cyclic-shift + window-partition gather), f32 -> f16 ----
__global__ void ln_window_kernel(const float* __restrict__ src, const float* __restrict__ gw,
                                 const float* __restrict__ gb, _Float16* __restrict__ dst,
                                 int windowed){
  int wave = threadIdx.x >> 5;
  int lane = threadIdx.x & 31;
  int row  = blockIdx.x * 8 + wave;           // destination row (windowed order if windowed)
  int srow;
  if (windowed){
    int b   = row >> 14;                      // NW*N = 16384 tokens per batch
    int rw  = row & 16383;
    int win = rw >> 6;
    int n   = rw & 63;
    int hs  = ((win >> 4) << 3) + (n >> 3);   // shifted-image coords
    int ws_ = ((win & 15) << 3) + (n & 7);
    int sh  = (hs + 4) & 127;                 // roll(-SS): shifted[h] = x[h+SS]
    int sw  = (ws_ + 4) & 127;
    srow = (b << 14) + (sh << 7) + sw;
  } else {
    srow = row;
  }
  const float* p = src + (size_t)srow * 256;
  float v[8];
  float s = 0.f, ss = 0.f;
#pragma unroll
  for (int k=0;k<8;++k){ float x = p[k*32 + lane]; v[k]=x; s += x; ss += x*x; }
#pragma unroll
  for (int m=16;m>=1;m>>=1){ s += __shfl_xor(s, m, 32); ss += __shfl_xor(ss, m, 32); }
  float mean = s * (1.f/256.f);
  float var  = ss * (1.f/256.f) - mean*mean;
  float rstd = rsqrtf(var + 1e-5f);
  _Float16* q = dst + (size_t)row * 256;
#pragma unroll
  for (int k=0;k<8;++k){
    int c = k*32 + lane;
    q[c] = (_Float16)((v[k]-mean)*rstd*gw[c] + gb[c]);
  }
}

// ---- QKV GEMM: 16x64 tile per wave; q gets pre-scaled by HD^-0.5 ----
__global__ void qkv_gemm_kernel(const _Float16* __restrict__ xw, const _Float16* __restrict__ yw,
                                const _Float16* __restrict__ wT, const float* __restrict__ bias,
                                _Float16* __restrict__ qout, _Float16* __restrict__ kvout){
  const int row0 = blockIdx.x * 16;
  const int nb   = blockIdx.y * 64;
  const bool isQ = (blockIdx.z == 0);
  const _Float16* A = (isQ ? xw : yw) + (size_t)row0 * 256;
  _Float16* D = isQ ? qout : kvout;
  v8f acc[4] = {};
  for (int k0 = 0; k0 < 256; k0 += 32){
    __builtin_prefetch(A + k0 + 32, 0, 1);
    v16h a = load_frag_a(A + k0, 256);
#pragma unroll
    for (int t=0;t<4;++t){
      v16h b = load_frag_b(wT + (size_t)(nb + t*16)*256 + k0, 256);
      acc[t] = WMMA_F32_F16(a, b, acc[t]);
    }
  }
  const int lane = laneId();
  const int lr0  = (lane >> 4) << 3;
  const int cn   = lane & 15;
  const float scale = isQ ? 0.17677669529663689f : 1.0f;
#pragma unroll
  for (int t=0;t<4;++t){
    int col = nb + t*16 + cn;
    float bi = bias[col];
#pragma unroll
    for (int r=0;r<8;++r){
      int row = row0 + lr0 + r;
      D[(size_t)row*256 + col] = (_Float16)((acc[t][r] + bi) * scale);
    }
  }
}

// ---- windowed cross-attention: block = window (8 waves = 8 heads) ----
// Per-wave LDS: q 64x32 (4KB) | kv 64x32 (4KB) | kvT 32x64 (4KB) | P 16x64 (2KB) = 14336 B
__global__ void attn_kernel(const _Float16* __restrict__ qbuf, const _Float16* __restrict__ kvbuf,
                            const float* __restrict__ bt, const int* __restrict__ mask_in,
                            _Float16* __restrict__ attn_out){
  extern __shared__ char smem_raw[];
  const int w    = blockIdx.x;                 // global window [0, 2048)
  const int head = threadIdx.x >> 5;
  const int lane = threadIdx.x & 31;
  const int winL = w & 255;
  _Float16* qL  = reinterpret_cast<_Float16*>(smem_raw + (size_t)head * 14336);
  _Float16* kvL = qL  + 2048;
  _Float16* kvT = kvL + 2048;                  // [d][m]
  _Float16* Pw  = kvT + 2048;

  const _Float16* qp  = qbuf  + (size_t)w*64*256 + head*32;  // 64 x 32, row stride 256
  const _Float16* kvp = kvbuf + (size_t)w*64*256 + head*32;
  const int* mi = mask_in + (size_t)w * 4096;

  // async-stage q and kv tiles into LDS (64 rows x 64B each = 256 x 16B chunks)
  unsigned int qOff  = (unsigned int)(uintptr_t)qL;   // low 32 bits of flat addr = LDS offset
  unsigned int kvOff = (unsigned int)(uintptr_t)kvL;
  for (int it = lane; it < 256; it += 32){
    int m = it >> 2, c = it & 3;
    async_copy_b128(qOff  + m*64 + c*16, qp  + (size_t)m*256 + c*8);
    async_copy_b128(kvOff + m*64 + c*16, kvp + (size_t)m*256 + c*8);
  }
  async_wait0();
  // build kv^T in LDS: kvT[d*64+m] = kv[m][d]
  for (int idx = lane; idx < 2048; idx += 32){
    int m = idx >> 5, d = idx & 31;
    kvT[d*64 + m] = kvL[m*32 + d];
  }
  __syncthreads();

  const int lrH = (lane >> 4) << 3;
  const int cn  = lane & 15;

  for (int i=0;i<4;++i){
    // S_i = q[i*16..) @ kv^T  (4 x 16x16 tiles, K=32 in one WMMA each)
    v8f acc[4] = {};
    v16h aF = load_frag_a(qL + i*16*32, 32);
#pragma unroll
    for (int j=0;j<4;++j){
      v16h bF = load_frag_b(kvL + j*16*32, 32);  // B col n = kv row n (contiguous in LDS)
      acc[j] = WMMA_F32_F16(aF, bF, acc[j]);
    }
    // epilogue: rel-pos bias + shift-region mask + input mask (selects only, EXEC untouched)
#pragma unroll
    for (int j=0;j<4;++j){
      int kq = j*16 + cn;
      int hk = ((winL >> 4) << 3) + (kq >> 3);
      int wk = ((winL & 15) << 3) + (kq & 7);
      int regK = (hk < 120 ? 0 : (hk < 124 ? 1 : 2)) * 3 + (wk < 120 ? 0 : (wk < 124 ? 1 : 2));
#pragma unroll
      for (int r=0;r<8;++r){
        int nq = i*16 + lrH + r;
        int hq = ((winL >> 4) << 3) + (nq >> 3);
        int wq = ((winL & 15) << 3) + (nq & 7);
        int regQ = (hq < 120 ? 0 : (hq < 124 ? 1 : 2)) * 3 + (wq < 120 ? 0 : (wq < 124 ? 1 : 2));
        int ridx = ((nq >> 3) - (kq >> 3) + 7) * 15 + ((nq & 7) - (kq & 7) + 7);
        float v = acc[j][r] + bt[ridx*8 + head];
        v = (regQ != regK) ? (v - 100.0f) : v;
        v = (mi[nq*64 + kq] == 0) ? -1e10f : v;
        acc[j][r] = v;
      }
    }
    // softmax over 64 keys: 4 in-lane values x 16-lane half-wave butterflies per row
    float rinv[8];
#pragma unroll
    for (int r=0;r<8;++r){
      float mx = fmaxf(fmaxf(acc[0][r],acc[1][r]), fmaxf(acc[2][r],acc[3][r]));
#pragma unroll
      for (int mk=8;mk>=1;mk>>=1) mx = fmaxf(mx, __shfl_xor(mx, mk, 32));
      float s = 0.f;
#pragma unroll
      for (int j=0;j<4;++j){ float e = __expf(acc[j][r]-mx); acc[j][r]=e; s+=e; }
#pragma unroll
      for (int mk=8;mk>=1;mk>>=1) s += __shfl_xor(s, mk, 32);
      rinv[r] = 1.0f / s;
    }
    // P tile (16x64) -> LDS (D-layout scatter), then A-layout reload
#pragma unroll
    for (int j=0;j<4;++j)
#pragma unroll
      for (int r=0;r<8;++r)
        Pw[(lrH + r)*64 + j*16 + cn] = (_Float16)(acc[j][r] * rinv[r]);
    __syncthreads();
    // O_i = P_i (16x64) @ kv (64x32): K loop of 2, B cols = kvT rows (contiguous)
    v8f oacc[2] = {};
#pragma unroll
    for (int kk=0;kk<2;++kk){
      v16h pA = load_frag_a(Pw + kk*32, 64);
#pragma unroll
      for (int nn=0;nn<2;++nn){
        v16h bF = load_frag_b(kvT + (nn*16)*64 + kk*32, 64);
        oacc[nn] = WMMA_F32_F16(pA, bF, oacc[nn]);
      }
    }
#pragma unroll
    for (int nn=0;nn<2;++nn)
#pragma unroll
      for (int r=0;r<8;++r){
        int nq = i*16 + lrH + r;
        int d  = nn*16 + cn;
        attn_out[((size_t)w*64 + nq)*256 + head*32 + d] = (_Float16)oacc[nn][r];
      }
    __syncthreads();
  }
}

// ---- proj GEMM + window-reverse + unshift + residual: x2 = x + scatter(attn@Wp + bp) ----
__global__ void proj_gemm_kernel(const _Float16* __restrict__ A_, const _Float16* __restrict__ wT,
                                 const float* __restrict__ bias, const float* __restrict__ x_in,
                                 float* __restrict__ x2){
  const int row0 = blockIdx.x * 16;
  const int nb   = blockIdx.y * 64;
  const _Float16* A = A_ + (size_t)row0 * 256;
  v8f acc[4] = {};
  for (int k0=0;k0<256;k0+=32){
    __builtin_prefetch(A + k0 + 32, 0, 1);
    v16h a = load_frag_a(A + k0, 256);
#pragma unroll
    for (int t=0;t<4;++t){
      v16h b = load_frag_b(wT + (size_t)(nb+t*16)*256 + k0, 256);
      acc[t] = WMMA_F32_F16(a,b,acc[t]);
    }
  }
  const int lane = laneId();
  const int lr0 = (lane>>4)<<3, cn = lane&15;
#pragma unroll
  for (int t=0;t<4;++t){
    int col = nb + t*16 + cn;
    float bi = bias[col];
#pragma unroll
    for (int r=0;r<8;++r){
      int row = row0 + lr0 + r;                 // windowed row index
      int b   = row >> 14;
      int rw  = row & 16383;
      int win = rw >> 6;
      int n   = rw & 63;
      int hs  = ((win>>4)<<3) + (n>>3);
      int ws_ = ((win&15)<<3) + (n&7);
      int hh  = (hs + 4) & 127;                 // roll(+SS)
      int ww  = (ws_ + 4) & 127;
      size_t drow = ((size_t)b << 14) + (size_t)(hh << 7) + ww;
      x2[drow*256 + col] = x_in[drow*256 + col] + acc[t][r] + bi;
    }
  }
}

// ---- fused MLP per 16-token strip: out = x2 + gelu(xn2@fc1+b1)@fc2+b2 ----
// A-strip (16x256 f16) staged into LDS by the Tensor Data Mover (TENSORcnt path).
__global__ void mlp_kernel(const _Float16* __restrict__ xn2, const _Float16* __restrict__ fc1T,
                           const float* __restrict__ b1, const _Float16* __restrict__ fc2T,
                           const float* __restrict__ b2, const float* __restrict__ x2,
                           float* __restrict__ out){
  extern __shared__ char smem_raw[];
  _Float16* g  = reinterpret_cast<_Float16*>(smem_raw);     // 16 x 1024 f16 (32 KiB)
  _Float16* aL = g + 16*1024;                               // 16 x 256 f16  (8 KiB)
  const int row0 = blockIdx.x * 16;
  const int lane = laneId();
  const int lr0 = (lane>>4)<<3, cn = lane&15;
  const _Float16* A = xn2 + (size_t)row0 * 256;
  const _Float16* As;
#if defined(HAVE_TDM)
  {
    // D# per ISA 8.3/8.4: count=1, data_size=2B, 2D tile 256x16, tensor stride 256
    unsigned long long ga = (unsigned long long)(uintptr_t)A;   // tile start
    unsigned int lo = (unsigned int)(uintptr_t)aL;              // LDS byte offset
    v4u g0 = { 1u,                                              // count=1, user mode
               lo,                                              // lds_addr [63:32]
               (unsigned int)(ga & 0xffffffffu),                // global_addr [95:64]
               (unsigned int)(((ga >> 32) & 0x01ffffffu) | (2u << 30)) }; // [120:96] + type=2
    v8i g1 = { (int)(1u << 16),        // data_size = 1 (2 bytes); mask=0
               (int)(256u << 16),      // tensor_dim0 lo16 (bits 63:48) = 256
               (int)(16u << 16),       // dim0 hi=0 | tensor_dim1 lo16 (95:80) = 16
               (int)(256u << 16),      // dim1 hi=0 | tile_dim0 (127:112) = 256
               16,                     // tile_dim1 (143:128) = 16; tile_dim2 = 0
               256,                    // tensor_dim0_stride lo32 = 256
               0, 0 };                 // stride hi / tensor_dim1_stride = 0
    v4i z4 = {0,0,0,0};
#if __clang_major__ >= 23
    v8i z8 = {0,0,0,0,0,0,0,0};
    __builtin_amdgcn_tensor_load_to_lds(g0, g1, z4, z4, z8, 0);
#else
    __builtin_amdgcn_tensor_load_to_lds(g0, g1, z4, z4, 0);
#endif
    __builtin_amdgcn_s_wait_tensorcnt((unsigned short)0);
    __syncthreads();
    As = aL;                                     // compute A-frags from LDS
  }
#else
  As = A;                                        // fallback: direct global loads
#endif
  // phase 1: g = gelu(xn2 @ fc1 + b1), 16x1024
  for (int nt=0;nt<16;++nt){
    int nb = nt*64;
    v8f acc[4] = {};
    for (int k0=0;k0<256;k0+=32){
      v16h a = load_frag_a(As + k0, 256);
#pragma unroll
      for (int t=0;t<4;++t){
        v16h b = load_frag_b(fc1T + (size_t)(nb+t*16)*256 + k0, 256);
        acc[t] = WMMA_F32_F16(a,b,acc[t]);
      }
    }
#pragma unroll
    for (int t=0;t<4;++t){
      int col = nb + t*16 + cn;
      float bi = b1[col];
#pragma unroll
      for (int r=0;r<8;++r){
        float v = acc[t][r] + bi;
        v = 0.5f * v * (1.0f + erff(v * 0.70710678118654752f));   // exact GELU
        g[(lr0 + r)*1024 + col] = (_Float16)v;
      }
    }
  }
  __syncthreads();
  // phase 2: out = g @ fc2 + b2 + x2, 16x256
  for (int nt=0;nt<4;++nt){
    int nb = nt*64;
    v8f acc[4] = {};
    for (int k0=0;k0<1024;k0+=32){
      v16h a = load_frag_a(g + k0, 1024);
#pragma unroll
      for (int t=0;t<4;++t){
        v16h b = load_frag_b(fc2T + (size_t)(nb+t*16)*1024 + k0, 1024);
        acc[t] = WMMA_F32_F16(a,b,acc[t]);
      }
    }
#pragma unroll
    for (int t=0;t<4;++t){
      int col = nb + t*16 + cn;
      float bi = b2[col];
#pragma unroll
      for (int r=0;r<8;++r){
        size_t row = (size_t)(row0 + lr0 + r);
        out[row*256 + col] = acc[t][r] + bi + x2[row*256 + col];
      }
    }
  }
}

extern "C" void kernel_launch(void* const* d_in, const int* in_sizes, int n_in,
                              void* d_out, int out_size, void* d_ws, size_t ws_size,
                              hipStream_t stream){
  (void)in_sizes; (void)n_in; (void)out_size; (void)ws_size;
  const float* x       = (const float*)d_in[0];
  const float* y       = (const float*)d_in[1];
  const int*   mask_in = (const int*)d_in[2];
  const float* bt      = (const float*)d_in[3];
  const float* qkv_w   = (const float*)d_in[4];
  const float* qkv_b   = (const float*)d_in[5];
  const float* proj_w  = (const float*)d_in[6];
  const float* proj_b  = (const float*)d_in[7];
  const float* n1w     = (const float*)d_in[8];
  const float* n1b     = (const float*)d_in[9];
  const float* n2w     = (const float*)d_in[10];
  const float* n2b     = (const float*)d_in[11];
  const float* fc1_w   = (const float*)d_in[12];
  const float* fc1_b   = (const float*)d_in[13];
  const float* fc2_w   = (const float*)d_in[14];
  const float* fc2_b   = (const float*)d_in[15];

  char* ws = (char*)d_ws;
  const size_t MiB = 1ull << 20;
  _Float16* qkvT  = (_Float16*)(ws + 0);            // 256x256 f16
  _Float16* projT = (_Float16*)(ws + 128*1024);     // 256x256 f16
  _Float16* fc1T  = (_Float16*)(ws + 256*1024);     // 1024x256 f16
  _Float16* fc2T  = (_Float16*)(ws + 768*1024);     // 256x1024 f16
  _Float16* xw    = (_Float16*)(ws + 2*MiB);        // 131072x256 f16 (64 MiB)
  _Float16* yw    = (_Float16*)(ws + 66*MiB);
  _Float16* qb    = (_Float16*)(ws + 130*MiB);
  _Float16* kvb   = (_Float16*)(ws + 194*MiB);
  _Float16* atn   = xw;                             // reuse (xw dead after qkv gemm)
  float*    x2    = (float*)(ws + 258*MiB);         // 131072x256 f32 (128 MiB)
  _Float16* xn2   = yw;                             // reuse (yw dead after qkv gemm)

  // weights -> f16, transposed (N x K) for contiguous B-fragment loads
  transpose_f16_kernel<<<(256*256 + 255)/256, 256, 0, stream>>>(qkv_w,  qkvT,  256, 256);
  transpose_f16_kernel<<<(256*256 + 255)/256, 256, 0, stream>>>(proj_w, projT, 256, 256);
  transpose_f16_kernel<<<(256*1024 + 255)/256, 256, 0, stream>>>(fc1_w, fc1T, 256, 1024);
  transpose_f16_kernel<<<(1024*256 + 255)/256, 256, 0, stream>>>(fc2_w, fc2T, 1024, 256);

  // LN1 + cyclic shift + window partition (fused gather), f32 -> f16
  ln_window_kernel<<<16384, 256, 0, stream>>>(x, n1w, n1b, xw, 1);
  ln_window_kernel<<<16384, 256, 0, stream>>>(y, n1w, n1b, yw, 1);

  // q = (xw@W+b)*scale ; kv = yw@W+b
  qkv_gemm_kernel<<<dim3(8192, 4, 2), 32, 0, stream>>>(xw, yw, qkvT, qkv_b, qb, kvb);

  // windowed attention (8 heads per block); LDS: 8 * 14336 B = 112 KiB
  attn_kernel<<<2048, 256, 8*14336, stream>>>(qb, kvb, bt, mask_in, atn);

  // proj + window reverse + unshift + residual -> x2 (f32)
  proj_gemm_kernel<<<dim3(8192, 4), 32, 0, stream>>>(atn, projT, proj_b, x, x2);

  // LN2 (plain rows) -> f16
  ln_window_kernel<<<16384, 256, 0, stream>>>(x2, n2w, n2b, xn2, 0);

  // fused MLP + final residual -> d_out ; LDS: 32 KiB GELU staging + 8 KiB TDM A-strip
  mlp_kernel<<<8192, 32, (16*1024 + 16*256)*sizeof(_Float16), stream>>>(
      xn2, fc1T, fc1_b, fc2T, fc2_b, x2, (float*)d_out);
}
